// HeadVQ_78417512890962
// MI455X (gfx1250) — matrix operand order repaired and checked by reading, not compiled, not run on hardware
//
#include <hip/hip_runtime.h>

#define KCODES 512
#define DDIM   128
#define TILE_M 256                        // 8 waves x 32 rows
#define NROWSQ (2 * 16 * 4096)            // 131072 rows per matrix
#define NTOT   ((size_t)NROWSQ * DDIM)    // 16777216 elems per mix output
#define SZB    136                        // bf16 LDS row stride (272B, 16B aligned)
#define CBHALF 256

typedef __attribute__((ext_vector_type(16))) __bf16 v16bf;
typedef __attribute__((ext_vector_type(8)))  __bf16 v8bf;
typedef __attribute__((ext_vector_type(4)))  __bf16 v4bf;
typedef __attribute__((ext_vector_type(8)))  float  v8f;

// Zero the scalar-loss + usage accumulators (4 + 2*512 floats) each launch.
__global__ __launch_bounds__(256) void vq_init_kernel(float* __restrict__ scal) {
    int t = blockIdx.x * blockDim.x + threadIdx.x;
    if (t < 4 + 2 * KCODES) scal[t] = 0.0f;
}

// Load one 16x32 bf16 A fragment (ISA layout: lanes 0-15 = M, lanes 16-31 same M
// with K offset; per-lane elems e<8 -> K = 8*half+e, e>=8 -> K = 8*half+e+8).
static __device__ __forceinline__ v16bf load_a_frag(const __bf16* __restrict__ z_s,
                                                    int row, int k0, int half) {
    const __bf16* zp = &z_s[row * SZB + k0 + half * 8];
    v8bf lo = *(const v8bf*)zp;
    v8bf hi = *(const v8bf*)(zp + 16);
    return __builtin_shufflevector(lo, hi,
        0, 1, 2, 3, 4, 5, 6, 7, 8, 9, 10, 11, 12, 13, 14, 15);
}

__global__ __launch_bounds__(256) void vq_quantize_kernel(
    const float* __restrict__ Kp, const float* __restrict__ Vp,
    const float* __restrict__ cbk, const float* __restrict__ cbv,
    float* __restrict__ out)
{
    __shared__ __bf16 z_s[TILE_M * SZB];    // 69632 B: block's z tile, bf16
    __shared__ __bf16 cb_s[CBHALF * SZB];   // 69632 B: half the codebook, bf16
    __shared__ float  c2_s[CBHALF];         // fp32 ||c||^2 for current half
    __shared__ int    idx_s[TILE_M];
    __shared__ float  red_s[8];

    const int t    = threadIdx.x;
    const int lane = t & 31;
    const int wave = t >> 5;
    const int hl   = lane & 15;            // lane within 16-lane half
    const int half = lane >> 4;            // 0 or 1

    const bool isV = (blockIdx.y != 0);
    const float* __restrict__ src = isV ? Vp : Kp;
    const float* __restrict__ cbg = isV ? cbv : cbk;
    float* __restrict__ mix   = out + (isV ? NTOT : (size_t)0);
    float* __restrict__ scal  = out + 2 * NTOT;
    float* __restrict__ usage = scal + 4 + (isV ? KCODES : 0);
    const int row0 = blockIdx.x * TILE_M;

    // ---- Stage z tile: fp32 global -> bf16 LDS (coalesced float4) ----
    {
        const float4* zg = (const float4*)(src + (size_t)row0 * DDIM);
        #pragma unroll 4
        for (int i = 0; i < 32; ++i) {
            int idx = i * 256 + t;            // 8192 float4 chunks
            int r = idx >> 5, c4 = idx & 31;  // row, float4-column
            float4 v = zg[idx];
            v4bf b;
            b[0] = (__bf16)v.x; b[1] = (__bf16)v.y;
            b[2] = (__bf16)v.z; b[3] = (__bf16)v.w;
            *(v4bf*)&z_s[r * SZB + c4 * 4] = b;
        }
    }
    __syncthreads();

    // ---- Hoist this wave's two 16-row A tiles into registers (64 VGPRs) ----
    v16bf a0[4], a1[4];
    #pragma unroll
    for (int kk = 0; kk < 4; ++kk) {
        a0[kk] = load_a_frag(z_s, wave * 32 + hl,      kk * 32, half);
        a1[kk] = load_a_frag(z_s, wave * 32 + 16 + hl, kk * 32, half);
    }

    float bestd0[8], bestd1[8];
    int   besti0[8], besti1[8];
    #pragma unroll
    for (int r = 0; r < 8; ++r) {
        bestd0[r] = 3.4e38f; besti0[r] = 0;
        bestd1[r] = 3.4e38f; besti1[r] = 0;
    }

    // ---- Loop over codebook in halves of 256 codes ----
    for (int hb = 0; hb < KCODES; hb += CBHALF) {
        __syncthreads();                       // protect cb_s / c2_s reuse
        if (t < CBHALF) c2_s[t] = 0.0f;
        __syncthreads();

        const float4* cg4 = (const float4*)(cbg + (size_t)hb * DDIM);
        #pragma unroll 4
        for (int i = 0; i < 32; ++i) {
            int idx = i * 256 + t;            // 8192 float4 chunks
            int r = idx >> 5, c4 = idx & 31;
            float4 v = cg4[idx];
            v4bf b;
            b[0] = (__bf16)v.x; b[1] = (__bf16)v.y;
            b[2] = (__bf16)v.z; b[3] = (__bf16)v.w;
            *(v4bf*)&cb_s[r * SZB + c4 * 4] = b;
            atomicAdd(&c2_s[r], v.x * v.x + v.y * v.y + v.z * v.z + v.w * v.w);
        }
        __syncthreads();

        // ---- 16-code tiles: each B fragment feeds TWO wmma (2 row tiles) ----
        for (int nt = 0; nt < CBHALF / 16; ++nt) {
            v8f acc0 = {}, acc1 = {};
            #pragma unroll
            for (int kk = 0; kk < 4; ++kk) {
                const int k0 = kk * 32;
                // B (32x16 bf16): lane = code; e<8 -> K = 8*half + e, e>=8 -> K = 8*half + e + 8
                const __bf16* bp = &cb_s[(nt * 16 + hl) * SZB + k0 + half * 8];
                v8bf blo = *(const v8bf*)bp;
                v8bf bhi = *(const v8bf*)(bp + 16);
                v16bf b = __builtin_shufflevector(blo, bhi,
                    0, 1, 2, 3, 4, 5, 6, 7, 8, 9, 10, 11, 12, 13, 14, 15);
                acc0 = __builtin_amdgcn_wmma_f32_16x16x32_bf16(
                    false, a0[kk], false, b, (short)0, acc0, false, false);
                acc1 = __builtin_amdgcn_wmma_f32_16x16x32_bf16(
                    false, a1[kk], false, b, (short)0, acc1, false, false);
            }
            // dist(m,n) = z2(m) + c2(n) - 2*logit; z2 constant per row -> drop
            float c2v  = c2_s[nt * 16 + hl];
            int   cidx = hb + nt * 16 + hl;
            #pragma unroll
            for (int r = 0; r < 8; ++r) {
                float d0 = c2v - 2.0f * acc0[r];   // row m=r+8*half of tile 0
                float d1 = c2v - 2.0f * acc1[r];   // row m=r+8*half of tile 1
                if (d0 < bestd0[r]) { bestd0[r] = d0; besti0[r] = cidx; }
                if (d1 < bestd1[r]) { bestd1[r] = d1; besti1[r] = cidx; }
            }
        }
    }

    // ---- Cross-lane argmin over the 16 codes-per-lane-group ----
    #pragma unroll
    for (int r = 0; r < 8; ++r) {
        float d0 = bestd0[r]; int i0 = besti0[r];
        float d1 = bestd1[r]; int i1 = besti1[r];
        #pragma unroll
        for (int off = 8; off >= 1; off >>= 1) {   // xor 8,4,2,1 stays in 16-lane half
            float od0 = __shfl_xor(d0, off, 32);
            int   oi0 = __shfl_xor(i0, off, 32);
            float od1 = __shfl_xor(d1, off, 32);
            int   oi1 = __shfl_xor(i1, off, 32);
            if (od0 < d0 || (od0 == d0 && oi0 < i0)) { d0 = od0; i0 = oi0; }
            if (od1 < d1 || (od1 == d1 && oi1 < i1)) { d1 = od1; i1 = oi1; }
        }
        if (hl == 0) {
            idx_s[wave * 32 + half * 8 + r]      = i0;
            idx_s[wave * 32 + 16 + half * 8 + r] = i1;
        }
    }
    __syncthreads();

    // ---- Output: gather fp32 codebook rows, fused MSE + usage ----
    float lsum = 0.0f;
    const float4* zg4 = (const float4*)(src + (size_t)row0 * DDIM);
    const float4* cg4 = (const float4*)cbg;
    float4*       og4 = (float4*)(mix + (size_t)row0 * DDIM);
    for (int j = wave; j < TILE_M; j += 8) {       // 32 rows per wave
        int gi = idx_s[j];
        float4 c = cg4[gi * 32 + lane];            // coalesced 16B/lane
        float4 z = zg4[j * 32 + lane];
        og4[j * 32 + lane] = c;                    // K_mix == z_q (eval gates=1)
        float dx = c.x - z.x, dy = c.y - z.y, dz = c.z - z.z, dw = c.w - z.w;
        lsum += dx * dx + dy * dy + dz * dz + dw * dw;
        if (lane == 0) atomicAdd(&usage[gi], 1.0f / (float)NROWSQ);
    }
    #pragma unroll
    for (int off = 16; off >= 1; off >>= 1) lsum += __shfl_xor(lsum, off, 32);
    if (lane == 0) red_s[wave] = lsum;
    __syncthreads();
    if (t == 0) {
        float s = 0.0f;
        #pragma unroll
        for (int w = 0; w < 8; ++w) s += red_s[w];
        float scaled = s * (0.25f / (float)NTOT); // BETA = GAMMA = 0.25
        atomicAdd(&scal[isV ? 1 : 0], scaled);    // BETA  * commit
        atomicAdd(&scal[isV ? 3 : 2], scaled);    // GAMMA * embed (same value)
    }
}

extern "C" void kernel_launch(void* const* d_in, const int* in_sizes, int n_in,
                              void* d_out, int out_size, void* d_ws, size_t ws_size,
                              hipStream_t stream) {
    const float* Kp  = (const float*)d_in[0];
    const float* Vp  = (const float*)d_in[1];
    const float* cbk = (const float*)d_in[2];
    const float* cbv = (const float*)d_in[3];
    float* out = (float*)d_out;

    vq_init_kernel<<<5, 256, 0, stream>>>(out + 2 * NTOT);

    dim3 grid(NROWSQ / TILE_M, 2);   // 512 row-tiles x {K, V}
    vq_quantize_kernel<<<grid, 256, 0, stream>>>(Kp, Vp, cbk, cbv, out);
}
